// QSAL_pennylane_32598801776765
// MI455X (gfx1250) — compile-verified, matrix operand order checked by reading, not run on hardware
//
#include <hip/hip_runtime.h>

#define NQ     9
#define NSTATE 512
#define BB     48
#define SS     96
#define NSAMP  (BB * SS)   // 4608
#define DIN    54
#define WDIM   90
#define DENC   4
#define DD     8
#define WAVES  4           // waves (samples) per workgroup

typedef __attribute__((ext_vector_type(16))) _Float16 v16h;
typedef __attribute__((ext_vector_type(8)))  float    v8f;

#define WB() __builtin_amdgcn_wave_barrier()

__device__ __forceinline__ float wave_reduce(float v) {
    #pragma unroll
    for (int off = 16; off > 0; off >>= 1) v += __shfl_xor(v, off, 32);
    return v;
}

// ---------------------------------------------------------------------------
// Gate passes. State: 512 complex (float2) in a wave-private LDS slice.
// qubit j <-> bit (8-j). All passes are wave-local: no workgroup barriers.
// ---------------------------------------------------------------------------

// Fused RX(t0) then RY(t1) on qubit q: complex 2x2 U = RY*RX.
__device__ __forceinline__ void rxry_pass(float2* dst, const float2* src, int lane, int q,
                                          float c0, float s0, float c1, float s1) {
    const int bit = 8 - q;
    const float u00r =  c1 * c0, u00i =  s1 * s0;
    const float u01r = -s1 * c0, u01i = -c1 * s0;
    const float u10r =  s1 * c0, u10i = -c1 * s0;
    const float u11r =  c1 * c0, u11i = -s1 * s0;
    #pragma unroll
    for (int t = 0; t < 8; ++t) {
        int p = lane + 32 * t;
        int low = p & ((1 << bit) - 1);
        int i0 = ((p >> bit) << (bit + 1)) | low;
        int i1 = i0 | (1 << bit);
        float2 a0 = src[i0], a1 = src[i1];
        dst[i0] = make_float2(u00r*a0.x - u00i*a0.y + u01r*a1.x - u01i*a1.y,
                              u00r*a0.y + u00i*a0.x + u01r*a1.y + u01i*a1.x);
        dst[i1] = make_float2(u10r*a0.x - u10i*a0.y + u11r*a1.x - u11i*a1.y,
                              u10r*a0.y + u10i*a0.x + u11r*a1.y + u11i*a1.x);
    }
}

// RY on two qubits (bit positions bh > bl). Optional permuted gather via Gt.
__device__ __forceinline__ void ry2_pass(float2* dst, const float2* src, int lane,
                                         int bh, int bl, float cH, float sH,
                                         float cL, float sL, const short* Gt) {
    #pragma unroll
    for (int t = 0; t < 4; ++t) {
        int g = lane + 32 * t;                       // 7 free bits
        int low = g & ((1 << bl) - 1);
        int mid = (g >> bl) & ((1 << (bh - bl - 1)) - 1);
        int hi  = g >> (bh - 1);
        int base = low | (mid << (bl + 1)) | (hi << (bh + 1));
        int i00 = base, i01 = base | (1 << bl);
        int i10 = base | (1 << bh), i11 = i10 | (1 << bl);
        float2 a00, a01, a10, a11;
        if (Gt) {
            a00 = src[Gt[i00]]; a01 = src[Gt[i01]];
            a10 = src[Gt[i10]]; a11 = src[Gt[i11]];
        } else {
            a00 = src[i00]; a01 = src[i01]; a10 = src[i10]; a11 = src[i11];
        }
        // RY on low qubit
        float2 t00 = make_float2(cL*a00.x - sL*a01.x, cL*a00.y - sL*a01.y);
        float2 t01 = make_float2(sL*a00.x + cL*a01.x, sL*a00.y + cL*a01.y);
        float2 t10 = make_float2(cL*a10.x - sL*a11.x, cL*a10.y - sL*a11.y);
        float2 t11 = make_float2(sL*a10.x + cL*a11.x, sL*a10.y + cL*a11.y);
        // RY on high qubit
        dst[i00] = make_float2(cH*t00.x - sH*t10.x, cH*t00.y - sH*t10.y);
        dst[i10] = make_float2(sH*t00.x + cH*t10.x, sH*t00.y + cH*t10.y);
        dst[i01] = make_float2(cH*t01.x - sH*t11.x, cH*t01.y - sH*t11.y);
        dst[i11] = make_float2(sH*t01.x + cH*t11.x, sH*t01.y + cH*t11.y);
    }
}

// RY on single qubit at bit position `bit`, in place.
__device__ __forceinline__ void ry1_pass(float2* buf, int lane, int bit, float c, float s) {
    #pragma unroll
    for (int t = 0; t < 8; ++t) {
        int p = lane + 32 * t;
        int low = p & ((1 << bit) - 1);
        int i0 = ((p >> bit) << (bit + 1)) | low;
        int i1 = i0 | (1 << bit);
        float2 a0 = buf[i0], a1 = buf[i1];
        buf[i0] = make_float2(c*a0.x - s*a1.x, c*a0.y - s*a1.y);
        buf[i1] = make_float2(s*a0.x + c*a1.x, s*a0.y + c*a1.y);
    }
}

// One entangler block: full 9-CNOT ring folded into the first RY pass as a
// permuted gather (src -> dst), then remaining RY pairs in place.
__device__ __forceinline__ void block_pass(float2* dst, const float2* src, int lane,
                                           const float* cv, const float* sv, int ab,
                                           const short* Gt) {
    ry2_pass(dst, src, lane, 8, 7, cv[ab+0], sv[ab+0], cv[ab+1], sv[ab+1], Gt); WB();
    ry2_pass(dst, dst, lane, 6, 5, cv[ab+2], sv[ab+2], cv[ab+3], sv[ab+3], nullptr); WB();
    ry2_pass(dst, dst, lane, 4, 3, cv[ab+4], sv[ab+4], cv[ab+5], sv[ab+5], nullptr); WB();
    ry2_pass(dst, dst, lane, 2, 1, cv[ab+6], sv[ab+6], cv[ab+7], sv[ab+7], nullptr); WB();
    ry1_pass(dst, lane, 0, cv[ab+8], sv[ab+8]); WB();
}

// Initial RX/RY layer (first pass copies from snapshot S) + nblk blocks.
// Returns buffer holding the final state (== A for even nblk).
__device__ __forceinline__ float2* run_circuit(float2* A, float2* Bf, const float2* S,
                                               int lane, const float* cv, const float* sv,
                                               int base, int nblk, const short* Gt) {
    rxry_pass(A, S, lane, 0, cv[base], sv[base], cv[base+1], sv[base+1]); WB();
    #pragma unroll
    for (int j = 1; j < NQ; ++j) {
        rxry_pass(A, A, lane, j, cv[base+2*j], sv[base+2*j],
                  cv[base+2*j+1], sv[base+2*j+1]); WB();
    }
    float2* cur = A; float2* alt = Bf;
    #pragma unroll 1
    for (int blk = 0; blk < nblk; ++blk) {
        block_pass(alt, cur, lane, cv, sv, base + 2*NQ + blk*NQ, Gt);
        float2* tmp = cur; cur = alt; alt = tmp;
    }
    return cur;
}

// ---------------------------------------------------------------------------
// Simulation kernel: one wave32 per sample; encoding stage reused for Q/K/V.
// ---------------------------------------------------------------------------
__global__ __launch_bounds__(32 * WAVES)
void qsal_sim_kernel(const float* __restrict__ inp,
                     const float* __restrict__ wq,
                     const float* __restrict__ wk,
                     const float* __restrict__ wv,
                     float* __restrict__ Qb, float* __restrict__ Kb,
                     float* __restrict__ Vb) {
    __shared__ float2 Ssh[WAVES][NSTATE];     // post-encoding snapshot
    __shared__ float2 Ash[WAVES][NSTATE];
    __shared__ float2 Bsh[WAVES][NSTATE];
    __shared__ float  cx[WAVES][DIN], sx[WAVES][DIN];   // per-sample x trig
    __shared__ float  cw[3 * WDIM], sw[3 * WDIM];       // shared weight trig
    __shared__ short  Gt[NSTATE];                       // CNOT-ring permutation

    const int tid = threadIdx.x;
    const int lane = tid & 31, wave = tid >> 5;
    const int m = blockIdx.x * WAVES + wave;            // flat sample = s*B + b
    const int s = m / BB, b = m % BB;
    const float* x = inp + (b * SS + s) * DIN;

    // Gather permutation of the 9-CNOT ring: new[k] = old[G(k)],
    // G = C0(C1(...(C8(k)))) with Cj: control bit 8-j, target bit (j<8 ? 7-j : 8).
    for (int i = tid; i < NSTATE; i += 32 * WAVES) {
        int k = i;
        #pragma unroll
        for (int j = NQ - 1; j >= 0; --j) {
            int bc = 8 - j;
            int bt = (j == 8) ? 8 : 7 - j;
            if ((k >> bc) & 1) k ^= (1 << bt);
        }
        Gt[i] = (short)k;
    }
    // shared weight trig (identical for all samples); fast HW transcendentals
    for (int t = tid; t < 3 * WDIM; t += 32 * WAVES) {
        int c = t / WDIM, k = t % WDIM;
        float a = (c == 0) ? wq[k] : (c == 1 ? wk[k] : wv[k]);
        float h = 0.5f * a;
        cw[t] = __cosf(h);
        sw[t] = __sinf(h);
    }
    // per-sample x trig (wave-local)
    for (int t = lane; t < DIN; t += 32) {
        float h = 0.5f * x[t];
        cx[wave][t] = __cosf(h);
        sx[wave][t] = __sinf(h);
    }
    __syncthreads();   // one-time: cw/sw and Gt are cross-wave

    float2* A  = Ash[wave];
    float2* Bf = Bsh[wave];
    float2* S  = Ssh[wave];

    // |0...0> init
    #pragma unroll
    for (int t = 0; t < 16; ++t) {
        int idx = lane + 32 * t;
        A[idx] = make_float2(idx == 0 ? 1.f : 0.f, 0.f);
    }
    WB();

    // encoding stage (x-dependent), ends in A (DENC even); snapshot to S
    run_circuit(A, Bf, A, lane, cx[wave], sx[wave], 0, DENC, Gt);
    #pragma unroll
    for (int t = 0; t < 16; ++t) { int idx = lane + 32 * t; S[idx] = A[idx]; }
    WB();

    #pragma unroll 1
    for (int c = 0; c < 3; ++c) {
        float2* cur = run_circuit(A, Bf, S, lane, cw, sw, c * WDIM, DD, Gt);

        if (c < 2) {
            float part = 0.f;
            #pragma unroll
            for (int t = 0; t < 16; ++t) {
                int idx = lane + 32 * t;
                float2 a = cur[idx];
                float p = a.x * a.x + a.y * a.y;
                part += (idx & 256) ? -p : p;           // <Z0>, qubit0 = bit8
            }
            float r = wave_reduce(part);
            if (lane == 0) (c == 0 ? Qb : Kb)[m] = r;
        } else {
            float* vo = Vb + m * DIN;
            // Z_q: one sweep over owned amplitudes, 9 accumulators
            float zp[NQ];
            #pragma unroll
            for (int q = 0; q < NQ; ++q) zp[q] = 0.f;
            #pragma unroll
            for (int t = 0; t < 16; ++t) {
                int idx = lane + 32 * t;
                float2 a = cur[idx];
                float p = a.x * a.x + a.y * a.y;
                #pragma unroll
                for (int q = 0; q < NQ; ++q)
                    zp[q] += ((idx >> (8 - q)) & 1) ? -p : p;
            }
            #pragma unroll
            for (int q = 0; q < NQ; ++q) {              // reduce+store immediately
                float r = wave_reduce(zp[q]);
                if (lane == 0) { vo[q] = r; vo[27 + q] = r; }
            }
            // X_q, Y_q: per qubit, reduce+store immediately (short live ranges)
            #pragma unroll 1
            for (int q = 0; q < NQ; ++q) {
                int bit = 8 - q;
                float xs = 0.f, ys = 0.f;
                #pragma unroll
                for (int t = 0; t < 8; ++t) {
                    int p2 = lane + 32 * t;
                    int low = p2 & ((1 << bit) - 1);
                    int i0 = ((p2 >> bit) << (bit + 1)) | low;
                    int i1 = i0 | (1 << bit);
                    float2 b0 = cur[i0], b1 = cur[i1];  // z = conj(a0)*a1
                    xs += 2.f * (b0.x * b1.x + b0.y * b1.y);
                    ys += 2.f * (b0.x * b1.y - b0.y * b1.x);
                }
                float rx = wave_reduce(xs);
                float ry = wave_reduce(ys);
                if (lane == 0) {
                    float* vq = vo + q;
                    vq[9]  = rx; vq[36] = rx;           // X block (g=1 and g=4)
                    vq[18] = ry; vq[45] = ry;           // Y block (g=2 and g=5)
                }
            }
        }
        WB();
    }
}

// ---------------------------------------------------------------------------
// Attention kernel: per-batch Gaussian-kernel attention, alpha@V via WMMA f16
// M=96 (6 tiles) x N=64 (4 tiles, 54 valid) x K=96 (3 steps of 32)
// ---------------------------------------------------------------------------
__global__ __launch_bounds__(256)
void qsal_attn_kernel(const float* __restrict__ inp,
                      const float* __restrict__ Qb, const float* __restrict__ Kb,
                      const float* __restrict__ Vb, float* __restrict__ outp) {
    __shared__ _Float16 Wh[SS * SS];     // normalized attention weights (96x96)
    __shared__ _Float16 Vh[SS * 64];     // V padded to 64 cols
    __shared__ float Qs[SS], Ks[SS];

    const int tid = threadIdx.x;
    const int b = blockIdx.x;
    const int wave = tid >> 5, lane = tid & 31;
    const int lm = lane & 15, half = lane >> 4;

    if (tid < SS) {
        Qs[tid] = Qb[tid * BB + b];
        Ks[tid] = Kb[tid * BB + b];
    }
    __syncthreads();

    if (tid < SS) {
        int i = tid;
        float qi = Qs[i], sum = 0.f;
        for (int j = 0; j < SS; ++j) { float dq = qi - Ks[j]; sum += __expf(-dq * dq); }
        float inv = 1.f / sum;
        for (int j = 0; j < SS; ++j) {
            float dq = qi - Ks[j];
            Wh[i * SS + j] = (_Float16)(__expf(-dq * dq) * inv);
        }
    }
    for (int idx = tid; idx < SS * 64; idx += 256) {
        int j = idx >> 6, cc = idx & 63;
        float v = (cc < DIN) ? Vb[(j * BB + b) * DIN + cc] : 0.f;
        Vh[idx] = (_Float16)v;
    }
    __syncthreads();

    for (int tile = wave; tile < 24; tile += 8) {
        int mt = tile >> 2, nt = tile & 3;
        v8f acc = {};
        int mrow = mt * 16 + lm;
        for (int kb = 0; kb < SS; kb += 32) {
            v16h A, Bm;
            #pragma unroll
            for (int v = 0; v < 8; ++v) {
                // ISA 16-bit A layout: lanes0-15 K={0..7,16..23}, lanes16-31 K={8..15,24..31}
                int kk = ((v & 4) ? 16 : 0) + (v & 3) * 2 + half * 8;
                A[2 * v]      = Wh[mrow * SS + kb + kk];
                A[2 * v + 1]  = Wh[mrow * SS + kb + kk + 1];
                Bm[2 * v]     = Vh[(kb + kk) * 64 + nt * 16 + lm];
                Bm[2 * v + 1] = Vh[(kb + kk + 1) * 64 + nt * 16 + lm];
            }
            acc = __builtin_amdgcn_wmma_f32_16x16x32_f16(
                false, A, false, Bm, (short)0, acc, false, false);
        }
        #pragma unroll
        for (int v = 0; v < 8; ++v) {
            int row = mt * 16 + v + half * 8;   // D layout: M = v + 8*half, N = lane&15
            int col = nt * 16 + lm;
            if (col < DIN) {
                int off = (b * SS + row) * DIN + col;
                outp[off] = inp[off] + acc[v];
            }
        }
    }
}

// ---------------------------------------------------------------------------
extern "C" void kernel_launch(void* const* d_in, const int* in_sizes, int n_in,
                              void* d_out, int out_size, void* d_ws, size_t ws_size,
                              hipStream_t stream) {
    const float* inp = (const float*)d_in[0];
    const float* wq  = (const float*)d_in[1];
    const float* wk  = (const float*)d_in[2];
    const float* wv  = (const float*)d_in[3];
    float* out = (float*)d_out;

    float* Qb = (float*)d_ws;            // 4608 floats
    float* Kb = Qb + NSAMP;              // 4608 floats
    float* Vb = Kb + NSAMP;              // 4608*54 floats (~1 MB scratch total)

    qsal_sim_kernel<<<NSAMP / WAVES, 32 * WAVES, 0, stream>>>(inp, wq, wk, wv, Qb, Kb, Vb);
    qsal_attn_kernel<<<BB, 256, 0, stream>>>(inp, Qb, Kb, Vb, out);
}
